// StableRNN_32641751450101
// MI455X (gfx1250) — matrix operand order repaired and checked by reading, not compiled, not run on hardware
//
#include <hip/hip_runtime.h>
#include <math.h>

// ---------------------------------------------------------------------------
// StableRNN on MI455X (gfx1250, wave32).
//   h_{t+1} = h_t + 0.1 * tanh(x[b,t]*U - h_t @ W^T),  W = exp(W_raw)
//   out[b,t] = c . h_{t+1}
// Latency-bound sequential scan (T=4096). 4 workgroups x 16 batches, 8 waves,
// wave w owns H-columns [16w,16w+16). Per step per wave: 2 independent
// 2-deep chains of v_wmma_f32_16x16x32_bf16 (recurrence) + 1 independent
// projection WMMA against Bc (c embedded in column 0). W and c resident in
// registers, h exchanged via double-buffered LDS A-fragments, one barrier
// per step, hardware V_TANH_F32. Branch-free hot loop (prologue/epilogue
// peeled), rotating output-fold wave.
// ---------------------------------------------------------------------------

typedef __attribute__((ext_vector_type(16))) __bf16 v16bf;
typedef __attribute__((ext_vector_type(8)))  float  v8f;

#define BSZ   64
#define TSZ   4096
#define HSZ   128
#define BT    16      // batch tile per workgroup
#define NWAVE 8
#define TC    64      // time-chunk staged in LDS
#define DT    0.1f

#define WFRAG_ELEMS (NWAVE * 4 * 32 * 16)   // 16384 bf16 (32 KB)

__device__ __forceinline__ unsigned short f32_to_bf16u(float f) {
    unsigned int u = __builtin_bit_cast(unsigned int, f);
    u += 0x7FFFu + ((u >> 16) & 1u);          // round-to-nearest-even
    return (unsigned short)(u >> 16);
}

// Hardware tanh (V_TANH_F32, single TRANS op) with branch-free fallback.
__device__ __forceinline__ float fast_tanh(float v) {
#if __has_builtin(__builtin_amdgcn_tanhf)
    return __builtin_amdgcn_tanhf(v);
#elif __has_builtin(__builtin_amdgcn_tanh_f32)
    return __builtin_amdgcn_tanh_f32(v);
#else
    float a = fabsf(v);
    float z = __expf(-2.0f * a);
    float r = __fdividef(1.0f - z, 1.0f + z);
    return copysignf(r, v);
#endif
}

// ---------------------------------------------------------------------------
// One-time prep.
// ids [0,1024): W = exp(W_raw) as per-wave B fragments Wfrag[w][kb][lane][e].
//   B layout (32x16 bf16): lane L -> N=L%16, K' = 16*(L/16)+e; k=32*kb+K',
//   n=16*w+N; value = W^T[k][n] = exp(W_raw[n*H+k]).
// ids [1024,1152): Bc fragments Cfrag[kb][lane][e]:
//   value = (n==0 ? c[k] : 0)  -> D column 0 = partial c.h over that kb.
// ---------------------------------------------------------------------------
__global__ void wprep_kernel(const float* __restrict__ W_raw,
                             const float* __restrict__ c,
                             unsigned short* __restrict__ frag) {
    int id = blockIdx.x * blockDim.x + threadIdx.x;
    int nW = NWAVE * 4 * 32;                  // 1024 fragment-rows for W
    int nC = 4 * 32;                          //  128 fragment-rows for Bc
    if (id >= nW + nC) return;
    unsigned short* dst = frag + (size_t)id * 16;
    if (id < nW) {
        int lane = id & 31;
        int kb   = (id >> 5) & 3;
        int w    = id >> 7;
        int g    = lane >> 4;
        int n    = w * 16 + (lane & 15);
        #pragma unroll
        for (int e = 0; e < 16; ++e) {
            int k = kb * 32 + g * 16 + e;
            dst[e] = f32_to_bf16u(expf(W_raw[n * HSZ + k]));
        }
    } else {
        int id2  = id - nW;
        int lane = id2 & 31;
        int kb   = id2 >> 5;
        int g    = lane >> 4;
        int n    = lane & 15;
        #pragma unroll
        for (int e = 0; e < 16; ++e) {
            int k = kb * 32 + g * 16 + e;
            dst[e] = (n == 0) ? f32_to_bf16u(c[k]) : (unsigned short)0;
        }
    }
}

// ---------------------------------------------------------------------------
// Main persistent scan kernel. grid = 4 blocks x 256 threads (8 waves).
// ---------------------------------------------------------------------------
__global__ __launch_bounds__(256, 1)
void stablernn_kernel(const float* __restrict__ x,      // [B][T]
                      const float* __restrict__ U,      // [H]
                      const unsigned short* __restrict__ frag, // W + Bc frags
                      float* __restrict__ out)          // [B][T]
{
    // A-fragment (16x32 bf16 per K-block) double buffer, WMMA A layout.
    __shared__ __align__(32) unsigned short Afrag[2][4][32][16];
    __shared__ float xbuf[BT][TC];
    __shared__ float outpart[2][4][BT];

    const int tid  = threadIdx.x;
    const int wave = tid >> 5;
    const int lane = tid & 31;
    const int g    = lane >> 4;          // half-wave group
    const int nloc = lane & 15;
    const int col  = wave * 16 + nloc;   // H column this lane owns in C/D
    const int b0   = blockIdx.x * BT;
    const int kbp  = wave & 3;           // K-block this wave projects

    // Resident W fragments: 4 x v16bf = 32 VGPRs; Bc fragment: 8 VGPRs.
    v16bf Bf[4];
    #pragma unroll
    for (int kb = 0; kb < 4; ++kb)
        Bf[kb] = *(const v16bf*)(frag + (((size_t)wave * 4 + kb) * 32 + lane) * 16);
    v16bf Bcf = *(const v16bf*)(frag + WFRAG_ELEMS +
                                (((size_t)kbp * 32) + lane) * 16);

    const float Ucoef = U[col];

    // Where this lane's h values land inside the A fragment:
    // A layout used here (contiguous k): element e, group gA -> K' = 16*gA+e
    //                                                     (matches B fragments).
    const int kbW = col >> 5;
    const int kpr = col & 31;
    const int gW  = kpr >> 4;
    const int eW  = kpr & 15;

    // ---- Prologue: stage x chunk 0, then h_1 = DT*tanh(x_0*U) directly ----
    for (int i = tid; i < BT * TC; i += 256) {
        int row = i >> 6;                  // i / TC
        int cc  = i & (TC - 1);
        xbuf[row][cc] = x[(size_t)(b0 + row) * TSZ + cc];
    }
    __syncthreads();

    float h[8];
    #pragma unroll
    for (int v = 0; v < 8; ++v) {
        int m = v + 8 * g;
        h[v] = DT * fast_tanh(xbuf[m][0] * Ucoef);
        Afrag[1][kbW][16 * gW + m][eW] = f32_to_bf16u(h[v]);
    }
    __syncthreads();

    // ---- Hot loop: iteration t consumes h_t (Afrag[t&1]), emits
    //      out[:,t-1] = c.h_t, produces h_{t+1} into Afrag[(t+1)&1]. ----
    for (int t = 1; t < TSZ; ++t) {
        const int cur = t & 1;
        const int nxt = cur ^ 1;
        const int tt  = t & (TC - 1);

        if (tt == 0) {
            // Stage next 64 time steps of x (coalesced), once per 64 steps.
            for (int i = tid; i < BT * TC; i += 256) {
                int row = i >> 6;
                int cc  = i & (TC - 1);
                xbuf[row][cc] = x[(size_t)(b0 + row) * TSZ + t + cc];
            }
            __syncthreads();
        }

        // Projection WMMA: partial c.h_t over K-block kbp (column 0 of D).
        {
            v16bf Ap = *(const v16bf*)(&Afrag[cur][kbp][lane][0]);
            v8f pz = {};
            pz = __builtin_amdgcn_wmma_f32_16x16x32_bf16(
                false, Ap, false, Bcf, (short)0, pz, false, false);
            if (wave < 4 && nloc == 0) {   // lanes 0 (m=0..7), 16 (m=8..15)
                #pragma unroll
                for (int v = 0; v < 8; ++v)
                    outpart[cur][wave][v + 8 * g] = pz[v];
            }
        }

        // y = h_t @ W^T: two independent 2-deep bf16 WMMA chains (K=128).
        v8f acc0 = {}, acc1 = {};
        {
            v16bf Af0 = *(const v16bf*)(&Afrag[cur][0][lane][0]);
            v16bf Af1 = *(const v16bf*)(&Afrag[cur][1][lane][0]);
            v16bf Af2 = *(const v16bf*)(&Afrag[cur][2][lane][0]);
            v16bf Af3 = *(const v16bf*)(&Afrag[cur][3][lane][0]);
            acc0 = __builtin_amdgcn_wmma_f32_16x16x32_bf16(
                false, Af0, false, Bf[0], (short)0, acc0, false, false);
            acc1 = __builtin_amdgcn_wmma_f32_16x16x32_bf16(
                false, Af2, false, Bf[2], (short)0, acc1, false, false);
            acc0 = __builtin_amdgcn_wmma_f32_16x16x32_bf16(
                false, Af1, false, Bf[1], (short)0, acc0, false, false);
            acc1 = __builtin_amdgcn_wmma_f32_16x16x32_bf16(
                false, Af3, false, Bf[3], (short)0, acc1, false, false);
        }
        v8f y = acc0 + acc1;

        // Elementwise update: lane holds rows m = v + 8*g, column `col`.
        #pragma unroll
        for (int v = 0; v < 8; ++v) {
            int m = v + 8 * g;
            float u = xbuf[m][tt] * Ucoef;
            h[v] += DT * fast_tanh(u - y[v]);
        }

        // Write h_{t+1} into next A buffer in fragment layout (8 x b16).
        #pragma unroll
        for (int v = 0; v < 8; ++v) {
            int m = v + 8 * g;
            Afrag[nxt][kbW][16 * gW + m][eW] = f32_to_bf16u(h[v]);
        }

        __syncthreads();   // single barrier per step (double-buffered state)

        // Rotating fold wave emits out[b][t-1] (off other waves' path).
        if (wave == (t & 7) && lane < BT) {
            float s = outpart[cur][0][lane] + outpart[cur][1][lane]
                    + outpart[cur][2][lane] + outpart[cur][3][lane];
            out[(size_t)(b0 + lane) * TSZ + (t - 1)] = s;
        }
    }

    // ---- Epilogue: out[:, TSZ-1] = c . h_TSZ  (h_TSZ is in Afrag[0]) ----
    __syncthreads();   // protect outpart against the last in-loop fold
    {
        v16bf Ap = *(const v16bf*)(&Afrag[0][kbp][lane][0]);
        v8f pz = {};
        pz = __builtin_amdgcn_wmma_f32_16x16x32_bf16(
            false, Ap, false, Bcf, (short)0, pz, false, false);
        if (wave < 4 && nloc == 0) {
            #pragma unroll
            for (int v = 0; v < 8; ++v)
                outpart[0][wave][v + 8 * g] = pz[v];
        }
        __syncthreads();
        if (wave == 0 && lane < BT) {
            float s = outpart[0][0][lane] + outpart[0][1][lane]
                    + outpart[0][2][lane] + outpart[0][3][lane];
            out[(size_t)(b0 + lane) * TSZ + (TSZ - 1)] = s;
        }
    }
}

// ---------------------------------------------------------------------------
extern "C" void kernel_launch(void* const* d_in, const int* in_sizes, int n_in,
                              void* d_out, int out_size, void* d_ws, size_t ws_size,
                              hipStream_t stream) {
    const float* x     = (const float*)d_in[0];   // [64,4096,1]
    const float* U     = (const float*)d_in[1];   // [128,1]
    const float* W_raw = (const float*)d_in[2];   // [128,128]
    const float* c     = (const float*)d_in[3];   // [1,128]
    float* out = (float*)d_out;                   // [64,4096,1]

    unsigned short* frag = (unsigned short*)d_ws; // 36 KB: W frags + Bc frags

    wprep_kernel<<<5, 256, 0, stream>>>(W_raw, c, frag);
    stablernn_kernel<<<4, 256, 0, stream>>>(x, U, frag, out);
}
